// testNet_7593502179527
// MI455X (gfx1250) — matrix operand order repaired and checked by reading, not compiled, not run on hardware
//
#include <hip/hip_runtime.h>

typedef __attribute__((ext_vector_type(16))) _Float16 v16h;
typedef __attribute__((ext_vector_type(8)))  _Float16 v8h;
typedef __attribute__((ext_vector_type(8)))  float    v8f;

namespace {
constexpr int kB = 16;      // batch
constexpr int kS = 1024;    // sequence length
constexpr int kL = 2048;    // attention context length
constexpr int kH = 32;      // LSTM hidden size
constexpr int kG = 128;     // 4*kH gate width
}

// Hardware tanh (V_TANH_F32 on gfx1250) with safe fallbacks.
#if __has_builtin(__builtin_amdgcn_tanhf)
__device__ __forceinline__ float fast_tanh(float x) { return __builtin_amdgcn_tanhf(x); }
#elif __has_builtin(__builtin_amdgcn_tanh_f32)
__device__ __forceinline__ float fast_tanh(float x) { return __builtin_amdgcn_tanh_f32(x); }
#else
__device__ __forceinline__ float fast_tanh(float x) { return tanhf(x); }
#endif

// ---------------------------------------------------------------------------
// Attention: softmax weights are independent of the query position (shift-
// invariance of softmax), so each (batch, stream) pair reduces to one
// softmax-weighted sum over 2048 value rows -> a single 3-vector.
// grid = 32 blocks (b,which), 256 threads.
// ---------------------------------------------------------------------------
__global__ void attn_kernel(const float* __restrict__ v1,
                            const float* __restrict__ v2,
                            const float* __restrict__ attn_w,
                            float* __restrict__ attn_out) {
    const int which = blockIdx.x & 1;
    const int b     = blockIdx.x >> 1;
    const float* v  = which ? v2 : v1;
    const int tid   = threadIdx.x;

    __shared__ float sc[kL];
    __shared__ float red[256 * 4];

    const float wk0 = attn_w[3], wk1 = attn_w[4], wk2 = attn_w[5];
    const float* vb = v + (size_t)b * (kL + 1) * 3;

    float mloc = -3.4e38f;
    for (int j = tid; j < kL; j += 256) {
        float s = vb[j*3+0]*wk0 + vb[j*3+1]*wk1 + vb[j*3+2]*wk2;
        sc[j] = s;
        mloc = fmaxf(mloc, s);
    }
    red[tid] = mloc;
    __syncthreads();
    for (int off = 128; off > 0; off >>= 1) {
        if (tid < off) red[tid] = fmaxf(red[tid], red[tid + off]);
        __syncthreads();
    }
    const float mx = red[0];
    __syncthreads();

    float a0 = 0.f, a1 = 0.f, a2 = 0.f, asum = 0.f;
    for (int j = tid; j < kL; j += 256) {
        float e = __expf(sc[j] - mx);
        asum += e;
        a0 += e * vb[(j+1)*3+0];
        a1 += e * vb[(j+1)*3+1];
        a2 += e * vb[(j+1)*3+2];
    }
    red[tid*4+0] = a0; red[tid*4+1] = a1; red[tid*4+2] = a2; red[tid*4+3] = asum;
    __syncthreads();
    for (int off = 128; off > 0; off >>= 1) {
        if (tid < off) {
            red[tid*4+0] += red[(tid+off)*4+0];
            red[tid*4+1] += red[(tid+off)*4+1];
            red[tid*4+2] += red[(tid+off)*4+2];
            red[tid*4+3] += red[(tid+off)*4+3];
        }
        __syncthreads();
    }
    if (tid == 0) {
        float inv = 1.0f / red[3];
        attn_out[which*48 + b*3 + 0] = red[0] * inv;
        attn_out[which*48 + b*3 + 1] = red[1] * inv;
        attn_out[which*48 + b*3 + 2] = red[2] * inv;
    }
}

// ---------------------------------------------------------------------------
// Precompute xg[s][gate_col][batch] = b_ih+b_hh + w_ih @ (x@fc1^T + fc1_b).
// grid = 1024 blocks (one per s), 256 threads.
// ---------------------------------------------------------------------------
__global__ void xg_kernel(const float* __restrict__ x,
                          const float* __restrict__ fc1_w, const float* __restrict__ fc1_b,
                          const float* __restrict__ w_ih,
                          const float* __restrict__ b_ih,  const float* __restrict__ b_hh,
                          float* __restrict__ xg) {
    const int s   = blockIdx.x;
    const int tid = threadIdx.x;
    __shared__ float lw[kG * 16];
    __shared__ float bsum[kG];
    __shared__ float xf[kB * 16];

    for (int i = tid; i < kG * 16; i += 256) lw[i] = w_ih[i];
    if (tid < kG) bsum[tid] = b_ih[tid] + b_hh[tid];

    {   // xf[b][k]: 256 threads == 16 batches * 16 features
        int b = tid >> 4, k = tid & 15;
        const float* xr = x + ((size_t)b * kS + s) * 3;
        xf[tid] = fc1_b[k] + xr[0]*fc1_w[k*3+0] + xr[1]*fc1_w[k*3+1] + xr[2]*fc1_w[k*3+2];
    }
    __syncthreads();

    const int n  = tid >> 1;            // gate column 0..127
    const int b0 = (tid & 1) * 8;       // batch half
    float* dst = xg + ((size_t)s * kG + n) * kB;
    for (int b = b0; b < b0 + 8; ++b) {
        float g = bsum[n];
        #pragma unroll
        for (int k = 0; k < 16; ++k) g += lw[n*16 + k] * xf[b*16 + k];
        dst[b] = g;
    }
}

// ---------------------------------------------------------------------------
// Persistent single-workgroup LSTM scan. 8 wave32s; each wave owns a 16-wide
// gate-column tile and does ONE v_wmma_f32_16x16x32_f16 per step:
//   gates(16x128) = h(16x32,f16) @ w_hh^T(32x128,f16), K=32 == hidden size.
// h lives in LDS as f16 row-major h[b][k]: each lane's A fragment is exactly
// two contiguous 16B LDS loads. Activations are branch-free: every gate is
// a*tanh(s*x)+b with wave-uniform (s,a,b) -> one V_TANH_F32 per element and
// no EXEC-mask divergence anywhere in the loop.
// ---------------------------------------------------------------------------
__global__ void __launch_bounds__(256, 1)
lstm_kernel(const float* __restrict__ xg,
            const float* __restrict__ w_hh,
            float* __restrict__ hs) {
    const int tid  = threadIdx.x;
    const int wave = tid >> 5;
    const int lane = tid & 31;

    __shared__ _Float16 hlds16[kB * kH];  // h[b][k], f16 (WMMA consumes f16 anyway)
    __shared__ float    act[kG * kB];     // activated gates [col][b]

    // B fragment (32x16 f16, K x N): lane -> column n = ncol; ISA layout:
    // lanes 0-15 hold K in {0..7, 16..23}, lanes 16-31 hold K in {8..15, 24..31}.
    const int ncol = (wave << 4) + (lane & 15);   // global gate column
    const int kbB  = (lane < 16) ? 0 : 8;
    v16h bfrag;
    #pragma unroll
    for (int h = 0; h < 16; ++h)
        bfrag[h] = (_Float16)w_hh[ncol * kH + kbB + (h < 8 ? h : h + 8)];  // B[k][n] = w_hh[n][k]

    hlds16[tid]       = (_Float16)0.0f;
    hlds16[tid + 256] = (_Float16)0.0f;
    float creg0 = 0.0f, creg1 = 0.0f;   // cell state: pairs p=tid, tid+256 (p = k*16 + b)

    // A fragment geometry (16x32 f16, M x K): lanes 0-15 row M=lane with
    // K in {0..7, 16..23}; lanes 16-31 row M=lane-16 with K in {8..15, 24..31}.
    const int m     = lane & 15;
    const int kbA   = (lane < 16) ? 0 : 8;
    const int rbase = (lane < 16) ? 0 : 8;   // C/D row base for this lane
    const _Float16* arow = &hlds16[m * kH + kbA];

    // Unified activation: gate = aco * tanh(sco * x) + bco
    //   sigmoid gates (i,f,o): (0.5, 0.5, 0.5);  cell gate g (waves 4,5): (1,1,0)
    const bool  is_g = ((wave >> 1) == 2);
    const float sco  = is_g ? 1.0f : 0.5f;
    const float aco  = is_g ? 1.0f : 0.5f;
    const float bco  = is_g ? 0.0f : 0.5f;

    __syncthreads();

    for (int s = 0; s < kS; ++s) {
        // two 16B LDS loads build the whole A fragment for this lane
        v8h alo = *(const v8h*)(arow);        // K = kbA .. kbA+7
        v8h ahi = *(const v8h*)(arow + 16);   // K = kbA+16 .. kbA+23
        v16h afrag;
        #pragma unroll
        for (int h = 0; h < 8; ++h) { afrag[h] = alo[h]; afrag[8 + h] = ahi[h]; }

        v8f cacc = {};
        cacc = __builtin_amdgcn_wmma_f32_16x16x32_f16(
            /*neg_a=*/false, afrag, /*neg_b=*/false, bfrag,
            /*c_mod=*/(short)0, cacc, /*reuse_a=*/false, /*reuse_b=*/false);

        // add precomputed input gates (two 16B loads) and activate branch-free
        const float* xgp = xg + ((size_t)s * kG + ncol) * kB + rbase;
        float4 xlo = *(const float4*)(xgp);
        float4 xhi = *(const float4*)(xgp + 4);
        float g[8];
        g[0] = cacc[0] + xlo.x; g[1] = cacc[1] + xlo.y;
        g[2] = cacc[2] + xlo.z; g[3] = cacc[3] + xlo.w;
        g[4] = cacc[4] + xhi.x; g[5] = cacc[5] + xhi.y;
        g[6] = cacc[6] + xhi.z; g[7] = cacc[7] + xhi.w;
        #pragma unroll
        for (int q = 0; q < 8; ++q)
            g[q] = fmaf(aco, fast_tanh(sco * g[q]), bco);

        float* ap = &act[ncol * kB + rbase];
        *(float4*)(ap)     = make_float4(g[0], g[1], g[2], g[3]);
        *(float4*)(ap + 4) = make_float4(g[4], g[5], g[6], g[7]);
        __syncthreads();

        // state update: 512 (k,b) pairs over 256 threads (2 each)
        {
            int k = tid >> 4, b = tid & 15;
            float iv = act[k*kB + b],          fv = act[(kH + k)*kB + b];
            float gv = act[(2*kH + k)*kB + b], ov = act[(3*kH + k)*kB + b];
            creg0 = fv * creg0 + iv * gv;
            float hv = ov * fast_tanh(creg0);
            hlds16[b*kH + k] = (_Float16)hv;
            hs[((size_t)b * kS + s) * kH + k] = hv;
        }
        {
            int p = tid + 256; int k = p >> 4, b = p & 15;
            float iv = act[k*kB + b],          fv = act[(kH + k)*kB + b];
            float gv = act[(2*kH + k)*kB + b], ov = act[(3*kH + k)*kB + b];
            creg1 = fv * creg1 + iv * gv;
            float hv = ov * fast_tanh(creg1);
            hlds16[b*kH + k] = (_Float16)hv;
            hs[((size_t)b * kS + s) * kH + k] = hv;
        }
        __syncthreads();
    }
}

// ---------------------------------------------------------------------------
// Head: out = relu([a1,a2,h] @ fc2^T + b2) @ fc3^T + b3. One thread per (b,s).
// ---------------------------------------------------------------------------
__global__ void head_kernel(const float* __restrict__ hs,
                            const float* __restrict__ attn_out,
                            const float* __restrict__ fc2_w, const float* __restrict__ fc2_b,
                            const float* __restrict__ fc3_w, const float* __restrict__ fc3_b,
                            float* __restrict__ out) {
    const int idx = blockIdx.x * blockDim.x + threadIdx.x;  // < 16384
    const int b = idx >> 10;
    const int s = idx & 1023;

    float cat[38];
    #pragma unroll
    for (int c = 0; c < 3; ++c) cat[c]     = attn_out[b*3 + c];        // a1[b]
    #pragma unroll
    for (int c = 0; c < 3; ++c) cat[3 + c] = attn_out[48 + b*3 + c];   // a2[b]
    const float* hp = hs + ((size_t)b * kS + s) * kH;
    #pragma unroll
    for (int k = 0; k < kH; ++k) cat[6 + k] = hp[k];

    float y[16];
    #pragma unroll
    for (int o = 0; o < 16; ++o) {
        float a = fc2_b[o];
        #pragma unroll
        for (int d = 0; d < 38; ++d) a += fc2_w[o*38 + d] * cat[d];
        y[o] = fmaxf(a, 0.0f);
    }
    float* op = out + (size_t)idx * 3;
    #pragma unroll
    for (int c = 0; c < 3; ++c) {
        float a = fc3_b[c];
        #pragma unroll
        for (int o = 0; o < 16; ++o) a += fc3_w[c*16 + o] * y[o];
        op[c] = a;
    }
}

extern "C" void kernel_launch(void* const* d_in, const int* in_sizes, int n_in,
                              void* d_out, int out_size, void* d_ws, size_t ws_size,
                              hipStream_t stream) {
    const float* x      = (const float*)d_in[0];
    const float* v1     = (const float*)d_in[1];
    const float* v2     = (const float*)d_in[2];
    const float* attn_w = (const float*)d_in[3];
    // d_in[4] attn_b: unused — softmax is shift-invariant, bias cancels exactly
    const float* fc1_w  = (const float*)d_in[5];
    const float* fc1_b  = (const float*)d_in[6];
    const float* w_ih   = (const float*)d_in[7];
    const float* w_hh   = (const float*)d_in[8];
    const float* b_ih   = (const float*)d_in[9];
    const float* b_hh   = (const float*)d_in[10];
    const float* fc2_w  = (const float*)d_in[11];
    const float* fc2_b  = (const float*)d_in[12];
    const float* fc3_w  = (const float*)d_in[13];
    const float* fc3_b  = (const float*)d_in[14];
    (void)in_sizes; (void)n_in; (void)out_size; (void)ws_size;

    char* ws = (char*)d_ws;
    float* xg   = (float*)(ws);                          // 1024*128*16 f32 = 8 MiB
    float* hsb  = (float*)(ws + (size_t)8  * 1024*1024); // 16*1024*32  f32 = 2 MiB
    float* attn = (float*)(ws + (size_t)10 * 1024*1024); // 2*16*3 f32

    xg_kernel  <<<kS,       256, 0, stream>>>(x, fc1_w, fc1_b, w_ih, b_ih, b_hh, xg);
    attn_kernel<<<2 * kB,   256, 0, stream>>>(v1, v2, attn_w, attn);
    lstm_kernel<<<1,        256, 0, stream>>>(xg, w_hh, hsb);
    head_kernel<<<kB*kS/256,256, 0, stream>>>(hsb, attn, fc2_w, fc2_b, fc3_w, fc3_b,
                                              (float*)d_out);
}
